// SparseDeepGCN_24515673326022
// MI455X (gfx1250) — matrix operand order repaired and checked by reading, not compiled, not run on hardware
//
#include <hip/hip_runtime.h>
#include <hip/hip_bf16.h>
#include <float.h>

// ---------------- types / helpers ----------------
typedef __bf16 bf16;
typedef bf16  v16bf __attribute__((ext_vector_type(16)));
typedef float v8f   __attribute__((ext_vector_type(8)));
typedef float v2f   __attribute__((ext_vector_type(2)));

static __device__ __forceinline__ bf16 to_bf16(float f) {
  // round-to-nearest-even f32 -> bf16
  unsigned u = __float_as_uint(f);
  unsigned r = (u + 0x7FFFu + ((u >> 16) & 1u)) >> 16;
  unsigned short s = (unsigned short)r;
  return __builtin_bit_cast(bf16, s);
}

static __device__ __forceinline__ v8f wmma_bf16(v16bf a, v16bf b, v8f c) {
  return __builtin_amdgcn_wmma_f32_16x16x32_bf16(false, a, false, b, (short)0, c, false, false);
}
static __device__ __forceinline__ v8f wmma_f32(v2f a, v2f b, v8f c) {
  return __builtin_amdgcn_wmma_f32_16x16x4_f32(false, a, false, b, (short)0, c, false, false);
}

// A fragment: 16x32 bf16 tile, rows row0..row0+15, k-slice k0..k0+31 (ISA 7.12.2 layout)
static __device__ __forceinline__ v16bf load_a_frag(const bf16* As, int lda, int row0, int k0, int lane) {
  int m = row0 + (lane & 15);
  int h = lane >> 4;
  v16bf a;
#pragma unroll
  for (int v = 0; v < 8; ++v) {
    int g  = v >> 2;
    int kk = k0 + g * 16 + h * 8 + (v & 3) * 2;
    a[2 * v]     = As[m * lda + kk];
    a[2 * v + 1] = As[m * lda + kk + 1];
  }
  return a;
}
// B fragment: 32x16 bf16 tile from row-major [K,N] LDS
static __device__ __forceinline__ v16bf load_b_frag(const bf16* Bs, int ldb, int k0, int col0, int lane) {
  int n = col0 + (lane & 15);
  int h = lane >> 4;
  v16bf b;
#pragma unroll
  for (int v = 0; v < 8; ++v) {
    int kk = k0 + (v >> 2) * 16 + h * 8 + (v & 3) * 2;
    b[2 * v]     = Bs[kk * ldb + n];
    b[2 * v + 1] = Bs[(kk + 1) * ldb + n];
  }
  return b;
}

#define ZERO8 {0.f,0.f,0.f,0.f,0.f,0.f,0.f,0.f}

// ---------------- tiny utility kernels ----------------
__global__ __launch_bounds__(256) void k_build_x(const float* pos, const float* color, float* x) {
  int n = blockIdx.x * 256 + threadIdx.x;
  if (n >= 8192) return;
  float* o = x + n * 9;
  o[0] = pos[n*3+0]; o[1] = pos[n*3+1]; o[2] = pos[n*3+2];
  o[3] = color[n*6+0]; o[4] = color[n*6+1]; o[5] = color[n*6+2];
  o[6] = color[n*6+3]; o[7] = color[n*6+4]; o[8] = color[n*6+5];
}

__global__ __launch_bounds__(256) void k_sqnorm(const float* X, int stride, int F, float* sq) {
  int n = blockIdx.x * 256 + threadIdx.x;
  if (n >= 8192) return;
  const float* r = X + (long)n * stride;
  float s = 0.f;
  for (int f = 0; f < F; ++f) s += r[f] * r[f];
  sq[n] = s;
}

// ---------------- distance matrix via fp32 WMMA ----------------
// dist[p][q] = sq[p] + sq[q] - 2 * <X_p, X_q> for one batch (2048x2048)
template <int KP>
__global__ __launch_bounds__(256) void k_dist(const float* X, int stride, int F,
                                              const float* sq, float* dist) {
  __shared__ float Xa[64 * KP];
  __shared__ float Xb[64 * KP];
  int tid = threadIdx.x, lane = tid & 31, wave = tid >> 5;
  int p0 = blockIdx.x * 64, q0 = blockIdx.y * 64;
  for (int e = tid; e < 64 * KP; e += 256) {
    int r = e / KP, k = e % KP;
    Xa[e] = (k < F) ? X[(long)(p0 + r) * stride + k] : 0.f;
    Xb[e] = (k < F) ? X[(long)(q0 + r) * stride + k] : 0.f;
  }
  __syncthreads();
  int h = lane >> 4, ln = lane & 15;
  for (int i = 0; i < 2; ++i) {
    int tt = wave * 2 + i, mt = tt >> 2, nt = tt & 3;
    v8f acc = ZERO8;
    for (int k0 = 0; k0 < KP; k0 += 4) {
      v2f a, b;
      int ka = (mt * 16 + ln) * KP + k0 + 2 * h;
      int kb = (nt * 16 + ln) * KP + k0 + 2 * h;
      a[0] = Xa[ka]; a[1] = Xa[ka + 1];
      b[0] = Xb[kb]; b[1] = Xb[kb + 1];
      acc = wmma_f32(a, b, acc);
    }
    int q = q0 + nt * 16 + ln;
    float sqq = sq[q];
#pragma unroll
    for (int v = 0; v < 8; ++v) {
      int p = p0 + mt * 16 + v + 8 * h;
      dist[(long)p * 2048 + q] = sq[p] + sqq - 2.0f * acc[v];
    }
  }
}

// ---------------- top-k (smallest, stable ties) with dilation ----------------
__global__ __launch_bounds__(256) void k_topk(const float* dist, int d, int* idxOut) {
  __shared__ float dv[2048];
  __shared__ float rv[256];
  __shared__ int   ri[256];
  int row = blockIdx.x, tid = threadIdx.x;
  for (int e = tid; e < 2048; e += 256) dv[e] = dist[(long)row * 2048 + e];
  __syncthreads();
  int m = 16 * d;
  for (int it = 0; it < m; ++it) {
    float best = FLT_MAX; int bi = 0x7fffffff;
    for (int e = tid; e < 2048; e += 256) {
      float v = dv[e];
      if (v < best) { best = v; bi = e; }
    }
    rv[tid] = best; ri[tid] = bi;
    __syncthreads();
    for (int s = 128; s > 0; s >>= 1) {
      if (tid < s) {
        float ov = rv[tid + s]; int oi = ri[tid + s];
        if (ov < rv[tid] || (ov == rv[tid] && oi < ri[tid])) { rv[tid] = ov; ri[tid] = oi; }
      }
      __syncthreads();
    }
    if (tid == 0) {
      if (it % d == 0) idxOut[row * 16 + it / d] = ri[0];
      dv[ri[0]] = FLT_MAX;
    }
    __syncthreads();
  }
}

// ---------------- EdgeConv: relu([x_i || x_j - x_i] W + b) max over j ----------------
// 8 points per workgroup -> 128 rows (point*16 + neighbor). One 16x16 M-tile == one point.
template <int KP>
__global__ __launch_bounds__(256) void k_edgeconv(const float* Xin, int stride, int F,
                                                  const int* idx, const float* Wg,
                                                  const float* bias, float* featsOut,
                                                  int outStride, int outCol, int doRes) {
  __shared__ bf16 As[128 * KP];
  __shared__ bf16 Ws[KP * 64];
  __shared__ float biasS[64];
  __shared__ int  qIdx[128];
  int tid = threadIdx.x, lane = tid & 31, wave = tid >> 5;
  int b = blockIdx.y;
  int p0 = blockIdx.x * 8;
  if (tid < 128) {
    int pi = tid >> 4, j = tid & 15;
    qIdx[tid] = idx[((long)(b * 2048 + p0 + pi)) * 16 + j];
  }
  if (tid < 64) biasS[tid] = bias[tid];
  __syncthreads();
  for (int e = tid; e < 128 * KP; e += 256) {
    int r = e / KP, k = e % KP;
    int p = p0 + (r >> 4);
    long ip = (long)(b * 2048 + p) * stride;
    float val;
    if (k < F) {
      val = Xin[ip + k];
    } else if (k < 2 * F) {
      int q = qIdx[r];
      long iq = (long)(b * 2048 + q) * stride;
      val = Xin[iq + (k - F)] - Xin[ip + (k - F)];
    } else {
      val = 0.f;
    }
    As[e] = to_bf16(val);
  }
  for (int e = tid; e < KP * 64; e += 256) {
    int k = e >> 6, n = e & 63;
    Ws[e] = (k < 2 * F) ? to_bf16(Wg[k * 64 + n]) : to_bf16(0.f);
  }
  __syncthreads();
  int h = lane >> 4, ln = lane & 15;
  for (int i = 0; i < 4; ++i) {
    int tt = wave * 4 + i, mt = tt >> 2, nt = tt & 3;
    v8f acc = ZERO8;
    for (int k0 = 0; k0 < KP; k0 += 32) {
      v16bf a = load_a_frag(As, KP, mt * 16, k0, lane);
      v16bf w = load_b_frag(Ws, 64, k0, nt * 16, lane);
      acc = wmma_bf16(a, w, acc);
    }
    float bn = biasS[nt * 16 + ln];
    float mx = 0.f;  // relu outputs are >= 0
#pragma unroll
    for (int v = 0; v < 8; ++v) {
      float y = acc[v] + bn;
      y = y > 0.f ? y : 0.f;
      mx = fmaxf(mx, y);
    }
    mx = fmaxf(mx, __shfl_xor(mx, 16, 32));  // combine the two row halves
    if (h == 0) {
      int p = p0 + mt;
      long orow = (long)(b * 2048 + p) * outStride + outCol + nt * 16 + ln;
      float res = doRes ? Xin[(long)(b * 2048 + p) * stride + nt * 16 + ln] : 0.f;
      featsOut[orow] = mx + res;
    }
  }
}

// ---------------- generic 64x64-tile bf16 WMMA GEMM ----------------
// AMODE 0: A = f32[M x lda] matrix.  AMODE 1: A row = [pooled(b) 1024 || feats(row) 448].
// EPI 0: out[row*Nout+col] = relu(acc + bias).  EPI 1: atomic max into pooled (no store).
template <int AMODE, int EPI>
__global__ __launch_bounds__(256) void k_gemm(const float* A, int lda, const float* W, int Nout,
                                              const float* bias, float* out, int Kdim,
                                              const float* pooled, const float* feats,
                                              unsigned* poolOut) {
  __shared__ bf16 As[64 * 32];
  __shared__ bf16 Bs[32 * 64];
  int tid = threadIdx.x, lane = tid & 31, wave = tid >> 5;
  int m0 = blockIdx.x * 64, n0 = blockIdx.y * 64;
  int h = lane >> 4, ln = lane & 15;
  int mt = wave & 3, nt0 = (wave >> 2) * 2;
  v8f acc0 = ZERO8, acc1 = ZERO8;
  for (int k0 = 0; k0 < Kdim; k0 += 32) {
    for (int e = tid; e < 64 * 32; e += 256) {
      int r = e >> 5, k = e & 31;
      int row = m0 + r, kk = k0 + k;
      float val;
      if constexpr (AMODE == 0) {
        val = A[(long)row * lda + kk];
      } else {
        val = (kk < 1024) ? pooled[(row >> 11) * 1024 + kk]
                          : feats[(long)row * 448 + (kk - 1024)];
      }
      As[e] = to_bf16(val);
    }
    for (int e = tid; e < 32 * 64; e += 256) {
      int k = e >> 6, n = e & 63;
      Bs[e] = to_bf16(W[(long)(k0 + k) * Nout + n0 + n]);
    }
    // prefetch next k-slice (global_prefetch_b8) while this tile computes
    if (k0 + 32 < Kdim) {
      int kk = k0 + 32;
      int row = m0 + (tid & 63);
      if constexpr (AMODE == 0) {
        __builtin_prefetch(&A[(long)row * lda + kk + (tid >> 6) * 8], 0, 1);
      } else {
        const float* pa = (kk < 1024) ? &pooled[(row >> 11) * 1024 + kk]
                                      : &feats[(long)row * 448 + (kk - 1024)];
        __builtin_prefetch(pa, 0, 1);
      }
      __builtin_prefetch(&W[(long)(kk + (tid >> 6) * 8) * Nout + n0 + (tid & 63)], 0, 1);
    }
    __syncthreads();
    v16bf a  = load_a_frag(As, 32, mt * 16, 0, lane);
    v16bf b0 = load_b_frag(Bs, 64, 0, nt0 * 16, lane);
    v16bf b1 = load_b_frag(Bs, 64, 0, (nt0 + 1) * 16, lane);
    acc0 = wmma_bf16(a, b0, acc0);
    acc1 = wmma_bf16(a, b1, acc1);
    __syncthreads();
  }
  for (int t = 0; t < 2; ++t) {
    v8f& acc = t ? acc1 : acc0;
    int nt = nt0 + t;
    int col = n0 + nt * 16 + ln;
    float bn = bias[col];
    if constexpr (EPI == 0) {
#pragma unroll
      for (int v = 0; v < 8; ++v) {
        int row = m0 + mt * 16 + v + 8 * h;
        float y = acc[v] + bn;
        y = y > 0.f ? y : 0.f;
        out[(long)row * Nout + col] = y;
      }
    } else {
      float mx = 0.f;
#pragma unroll
      for (int v = 0; v < 8; ++v) {
        float y = acc[v] + bn;
        y = y > 0.f ? y : 0.f;
        mx = fmaxf(mx, y);
      }
      mx = fmaxf(mx, __shfl_xor(mx, 16, 32));
      if (h == 0) {
        int bidx = (m0 + mt * 16) >> 11;  // tiles never straddle a batch (2048 % 64 == 0)
        atomicMax(&poolOut[bidx * 1024 + col], __float_as_uint(mx));  // values >= 0
      }
    }
  }
}

// ---------------- final small projection (N=13) ----------------
__global__ __launch_bounds__(256) void k_p3(const float* h2, const float* W, const float* bias,
                                            float* out) {
  int t = blockIdx.x * 256 + threadIdx.x;
  if (t >= 8192 * 13) return;
  int row = t / 13, c = t % 13;
  float s = bias[c];
  const float* hr = h2 + (long)row * 256;
  for (int k = 0; k < 256; ++k) s += hr[k] * W[k * 13 + c];
  out[t] = s;
}

// ---------------- host orchestration ----------------
extern "C" void kernel_launch(void* const* d_in, const int* in_sizes, int n_in,
                              void* d_out, int out_size, void* d_ws, size_t ws_size,
                              hipStream_t stream) {
  (void)in_sizes; (void)n_in; (void)out_size; (void)ws_size;
  const float* pos      = (const float*)d_in[0];
  const float* color    = (const float*)d_in[1];
  const float* W_head   = (const float*)d_in[2];
  const float* b_head   = (const float*)d_in[3];
  const float* W_blocks = (const float*)d_in[4];
  const float* b_blocks = (const float*)d_in[5];
  const float* W_f      = (const float*)d_in[6];
  const float* b_f      = (const float*)d_in[7];
  const float* W_p1     = (const float*)d_in[8];
  const float* b_p1     = (const float*)d_in[9];
  const float* W_p2     = (const float*)d_in[10];
  const float* b_p2     = (const float*)d_in[11];
  const float* W_p3     = (const float*)d_in[12];
  const float* b_p3     = (const float*)d_in[13];
  // d_in[14] = batch_idx (implicit: row >> 11)

  char* ws = (char*)d_ws;
  size_t off = 0;
  auto alloc = [&](size_t bytes) -> void* {
    void* p = ws + off;
    off = (off + bytes + 255) & ~(size_t)255;
    return p;
  };
  float* x      = (float*)alloc((size_t)8192 * 9 * 4);
  float* feats  = (float*)alloc((size_t)8192 * 448 * 4);
  float* sq     = (float*)alloc((size_t)8192 * 4);
  float* dist   = (float*)alloc((size_t)2048 * 2048 * 4);
  int*   idx    = (int*)alloc((size_t)4 * 2048 * 16 * 4);
  float* pooled = (float*)alloc((size_t)4 * 1024 * 4);
  float* h1     = (float*)alloc((size_t)8192 * 512 * 4);
  float* h2     = (float*)alloc((size_t)8192 * 256 * 4);

  k_build_x<<<32, 256, 0, stream>>>(pos, color, x);

  // head: kNN on xyz only (F=3), dilation 1
  k_sqnorm<<<32, 256, 0, stream>>>(x, 9, 3, sq);
  for (int b = 0; b < 4; ++b) {
    k_dist<4><<<dim3(32, 32), 256, 0, stream>>>(x + (size_t)b * 2048 * 9, 9, 3,
                                                sq + b * 2048, dist);
    k_topk<<<2048, 256, 0, stream>>>(dist, 1, idx + b * 2048 * 16);
  }
  k_edgeconv<32><<<dim3(256, 4), 256, 0, stream>>>(x, 9, 9, idx, W_head, b_head,
                                                   feats, 448, 0, 0);

  // 6 ResDynBlocks: kNN on current features, dilation 1+i, residual add
  for (int i = 0; i < 6; ++i) {
    const float* Xi = feats + i * 64;
    k_sqnorm<<<32, 256, 0, stream>>>(Xi, 448, 64, sq);
    for (int b = 0; b < 4; ++b) {
      k_dist<64><<<dim3(32, 32), 256, 0, stream>>>(Xi + (size_t)b * 2048 * 448, 448, 64,
                                                   sq + b * 2048, dist);
      k_topk<<<2048, 256, 0, stream>>>(dist, i + 1, idx + b * 2048 * 16);
    }
    k_edgeconv<128><<<dim3(256, 4), 256, 0, stream>>>(Xi, 448, 64, idx,
                                                      W_blocks + (size_t)i * 128 * 64,
                                                      b_blocks + i * 64, feats, 448,
                                                      (i + 1) * 64, 1);
  }

  // fusion GEMM with fused global max-pool epilogue (no [N,1024] materialization)
  hipMemsetAsync(pooled, 0, (size_t)4 * 1024 * 4, stream);
  k_gemm<0, 1><<<dim3(128, 16), 256, 0, stream>>>(feats, 448, W_f, 1024, b_f, nullptr, 448,
                                                  nullptr, nullptr, (unsigned*)pooled);
  // p1: A = [pooled(b) || feats(row)] built on the fly
  k_gemm<1, 0><<<dim3(128, 8), 256, 0, stream>>>(nullptr, 0, W_p1, 512, b_p1, h1, 1472,
                                                 pooled, feats, nullptr);
  // p2
  k_gemm<0, 0><<<dim3(128, 4), 256, 0, stream>>>(h1, 512, W_p2, 256, b_p2, h2, 512,
                                                 nullptr, nullptr, nullptr);
  // p3 -> d_out [8192,13] f32
  k_p3<<<(8192 * 13 + 255) / 256, 256, 0, stream>>>(h2, W_p3, b_p3, (float*)d_out);
}
